// WindowedStructCmapCATT_10376640987598
// MI455X (gfx1250) — matrix operand order repaired and checked by reading, not compiled, not run on hardware
//
#include <hip/hip_runtime.h>
#include <hip/hip_bf16.h>

// ---------------------------------------------------------------------------
// Problem constants (from reference): B=8, L=512, INIT_DIM=1024, D=512, H=8,
// NBINS=25, PW=10, WSIZE=5, HALF=256.
// ---------------------------------------------------------------------------
#define Bc     8
#define Lc     512
#define INc    1024
#define Dc     512
#define Hc     8
#define DHc    64        // D / H
#define TDc    1536      // 3*D
#define NBc    25
#define HALFc  256
#define MBLc   4096      // B*L
#define PWc    10

typedef __attribute__((ext_vector_type(16))) _Float16 v16h;
typedef __attribute__((ext_vector_type(8)))  float    v8f;
typedef __attribute__((ext_vector_type(4)))  unsigned u32x4;

// Batch (z) -> element-offset map: off = (z/div)*s1 + (z%mod)*s2
struct ZMap { int div; long long s1; int mod; long long s2; };

__device__ __forceinline__ size_t zoff(const ZMap m, int z) {
  return (size_t)(z / m.div) * (size_t)m.s1 + (size_t)(z % m.mod) * (size_t)m.s2;
}

// ---------------------------------------------------------------------------
// Generic batched WMMA GEMM:
//   out = act( scale * (A @ B^T) + bias + resid )
// A: f16 [M,K] row-major (lda). B: f16 [N,K] row-major (ldb) -- always the
// "weight transpose" orientation; operands needing [K,N] are pre-transposed
// on the host side (Wbins cast-transpose, V via vtrans_kernel).
// Each wave computes a 2x2 grid of 16x16 tiles (32x32 output): A fragments
// are reused across two B fragments and vice versa -> 2x arithmetic
// intensity, 4 WMMAs per K-step (8 with unroll-2).
// Block = 256 threads = 8 waves arranged 2(M) x 4(N): block tile 64x128.
// Fragment layouts per CDNA5 ISA 7.12.2 (wave32):
//   A 16x32 f16: lanes 0-15 half=0, 16-31 half=1; VGPR pairs cover
//     halfs [k+8h .. k+8h+7] and [k+16+8h .. k+16+8h+7]  (two b128 loads)
//   B 32x16 f16: lane n = lane&15; halfs [k+8h .. k+8h+7] and [k+16+8h .. k+16+8h+7] (two b128 loads)
//   C/D 16x16 f32: VGPR g -> M = g + 8*half, N = lane&15
// ---------------------------------------------------------------------------
__global__ __launch_bounds__(256)
void wmma_gemm_kernel(const _Float16* __restrict__ A,
                      const _Float16* __restrict__ Bm,
                      const float*    __restrict__ bias,
                      const float*    __restrict__ resid,
                      float*          __restrict__ outF,
                      _Float16*       __restrict__ outH,
                      int M, int N, int K, int lda, int ldb, int ldc,
                      ZMap za, ZMap zb, ZMap zc,
                      float scale, int actF, int actH)
{
  const int wave  = threadIdx.x >> 5;
  const int lane  = threadIdx.x & 31;
  const int waveM = wave >> 2;                       // 0..1
  const int waveN = wave & 3;                        // 0..3
  const int tileM  = blockIdx.y * 64  + waveM * 32;  // rows tileM, tileM+16
  const int tileN0 = blockIdx.x * 128 + waveN * 32;  // cols tileN0, tileN0+16
  if (tileM >= M || tileN0 >= N) return;             // wave-uniform guards
  const bool n1ok = (tileN0 + 16) < N;               // wave-uniform
  const int z = blockIdx.z;

  const _Float16* Ab = A  + zoff(za, z);
  const _Float16* Bb = Bm + zoff(zb, z);
  const size_t cbase = zoff(zc, z);

  const int nlo  = lane & 15;
  const int half = lane >> 4;

  const _Float16* arow0 = Ab + (size_t)(tileM  + nlo) * lda + 8 * half;
  const _Float16* arow1 = arow0 + (size_t)16 * lda;
  const _Float16* brow0 = Bb + (size_t)(tileN0 + nlo) * ldb + 8 * half;
  const _Float16* brow1 = brow0 + (size_t)16 * ldb;

  union Frag { v16h v; u32x4 q[2]; };
  v8f acc00 = {}, acc01 = {}, acc10 = {}, acc11 = {};

  #pragma unroll 2
  for (int k = 0; k < K; k += 32) {
    __builtin_prefetch((const void*)(arow0 + k + 128), 0, 1); // global_prefetch_b8
    Frag a0, a1, b0, b1;
    a0.q[0] = *(const u32x4*)(arow0 + k);
    a0.q[1] = *(const u32x4*)(arow0 + k + 16);
    a1.q[0] = *(const u32x4*)(arow1 + k);
    a1.q[1] = *(const u32x4*)(arow1 + k + 16);
    b0.q[0] = *(const u32x4*)(brow0 + k);
    b0.q[1] = *(const u32x4*)(brow0 + k + 16);
    acc00 = __builtin_amdgcn_wmma_f32_16x16x32_f16(false, a0.v, false, b0.v,
                                                   (short)0, acc00, false, false);
    acc10 = __builtin_amdgcn_wmma_f32_16x16x32_f16(false, a1.v, false, b0.v,
                                                   (short)0, acc10, false, false);
    if (n1ok) {
      b1.q[0] = *(const u32x4*)(brow1 + k);
      b1.q[1] = *(const u32x4*)(brow1 + k + 16);
      acc01 = __builtin_amdgcn_wmma_f32_16x16x32_f16(false, a0.v, false, b1.v,
                                                     (short)0, acc01, false, false);
      acc11 = __builtin_amdgcn_wmma_f32_16x16x32_f16(false, a1.v, false, b1.v,
                                                     (short)0, acc11, false, false);
    }
  }

  auto store_tile = [&](const v8f& acc, int rBase, int cBase) {
    const int col = cBase + nlo;
    const float bv = bias ? bias[col] : 0.0f;
    #pragma unroll
    for (int g = 0; g < 8; ++g) {
      const int row = rBase + g + 8 * half;
      const size_t idx = cbase + (size_t)row * ldc + col;
      float v = acc[g] * scale + bv;
      if (resid) v += resid[idx];
      if (outF)  outF[idx] = actF ? tanhf(v) : v;
      if (outH)  outH[idx] = (_Float16)(actH ? tanhf(v) : v);
    }
  };
  store_tile(acc00, tileM,      tileN0);
  store_tile(acc10, tileM + 16, tileN0);
  if (n1ok) {
    store_tile(acc01, tileM,      tileN0 + 16);
    store_tile(acc11, tileM + 16, tileN0 + 16);
  }
}

// ---------------------------------------------------------------------------
// f32 -> f16 elementwise cast
// ---------------------------------------------------------------------------
__global__ void cvt_f32_f16_kernel(const float* __restrict__ src,
                                   _Float16* __restrict__ dst, size_t n)
{
  size_t i = (size_t)blockIdx.x * blockDim.x + threadIdx.x;
  size_t stride = (size_t)gridDim.x * blockDim.x;
  for (; i < n; i += stride) dst[i] = (_Float16)src[i];
}

// ---------------------------------------------------------------------------
// Cast + transpose Wbins: dst[n][kk][d] = (f16)src[n][d][kk]
// Puts the bins weights into [N,K] orientation for the WMMA GEMM.
// ---------------------------------------------------------------------------
__global__ void wbins_transpose_kernel(const float* __restrict__ src,
                                       _Float16* __restrict__ dst)
{
  size_t idx = (size_t)blockIdx.x * blockDim.x + threadIdx.x;
  if (idx >= (size_t)NBc * Dc * HALFc) return;
  const int d  = (int)(idx % Dc);
  const size_t r = idx / Dc;
  const int kk = (int)(r % HALFc);
  const int n  = (int)(r / HALFc);
  dst[idx] = (_Float16)src[((size_t)n * Dc + d) * HALFc + kk];
}

// ---------------------------------------------------------------------------
// Transpose the V slice of a packed QKV buffer into vt[(b*H+h)*DH + d][l]
// so P@V can use the coalesced [N,K] GEMM path.
// One block per (b,h, 32-l x 64-d tile). Global -> LDS via the CDNA5 async
// LDS-load path (ASYNCcnt), then transposed ds reads -> coalesced stores.
// ---------------------------------------------------------------------------
__global__ __launch_bounds__(256)
void vtrans_kernel(const _Float16* __restrict__ qkvV,   // = qkv + 2*D
                   _Float16* __restrict__ vt)
{
  __shared__ _Float16 tile[32][72];   // 64 cols + 8 pad halfs (bank stagger)
  const int t  = threadIdx.x;
  const int bh = blockIdx.x >> 4;                 // 0..63 = b*H+h
  const int lt = (blockIdx.x & 15) * 32;          // l tile base
  const int b  = bh >> 3, h = bh & 7;

  // Stage 32x64 halfs (1024 dwords): 4 async b32 loads per thread.
  #pragma unroll
  for (int i = 0; i < 4; ++i) {
    const int e     = t + i * 256;                // dword index 0..1023
    const int l     = e >> 5;                     // 0..31
    const int dpair = e & 31;                     // dword within row
    const _Float16* src =
        qkvV + ((size_t)b * Lc + lt + l) * TDc + h * DHc + dpair * 2;
    // Generic LDS pointer: low 32 bits are the LDS byte offset.
    const unsigned ldsOff = (unsigned)(size_t)&tile[l][dpair * 2];
    asm volatile("global_load_async_to_lds_b32 %0, %1, off"
                 :: "v"(ldsOff), "v"(src) : "memory");
  }
  asm volatile("s_wait_asynccnt 0x0" ::: "memory");
  __syncthreads();

  // Write transposed: vt[(bh*64 + d)*L + lt + l], packing two l's per dword.
  #pragma unroll
  for (int i = 0; i < 4; ++i) {
    const int e  = t + i * 256;                   // 0..1023
    const int d  = e >> 4;                        // 0..63
    const int lp = e & 15;                        // l-pair, l = 2*lp
    _Float16* dst = vt + ((size_t)bh * DHc + d) * Lc + lt + 2 * lp;
    dst[0] = tile[2 * lp][d];
    dst[1] = tile[2 * lp + 1][d];
  }
}

// ---------------------------------------------------------------------------
// Row softmax, one wave32 per row (rows of length 512); f32 in, f16 out.
// ---------------------------------------------------------------------------
__global__ __launch_bounds__(256)
void softmax_rows_kernel(const float* __restrict__ S, _Float16* __restrict__ P,
                         int rows, int n)
{
  const int wave = threadIdx.x >> 5;
  const int lane = threadIdx.x & 31;
  const int row  = blockIdx.x * 8 + wave;
  if (row >= rows) return;
  const float* s = S + (size_t)row * n;
  float mx = -3.4e38f;
  for (int i = lane; i < n; i += 32) mx = fmaxf(mx, s[i]);
  #pragma unroll
  for (int o = 16; o; o >>= 1) mx = fmaxf(mx, __shfl_xor(mx, o, 32));
  float sum = 0.0f;
  for (int i = lane; i < n; i += 32) sum += __expf(s[i] - mx);
  #pragma unroll
  for (int o = 16; o; o >>= 1) sum += __shfl_xor(sum, o, 32);
  const float inv = 1.0f / sum;
  _Float16* p = P + (size_t)row * n;
  for (int i = lane; i < n; i += 32) p[i] = (_Float16)(__expf(s[i] - mx) * inv);
}

// ---------------------------------------------------------------------------
// Fused bin smoothing (triangular window over NBINS axis, zero pad 2) +
// agg-weighted bin reduction + tanh. One thread per (b, i, j): keeps the 25
// bin values in registers -> 1 read + 1 write of the 210MB tensor.
// ---------------------------------------------------------------------------
__global__ void smooth_agg_kernel(const float* __restrict__ cmpre,
                                  const float* __restrict__ smooth,
                                  const float* __restrict__ agg,
                                  float* __restrict__ cmOut,
                                  float* __restrict__ cm1)
{
  const size_t PLANE = (size_t)Lc * Lc;
  size_t idx = (size_t)blockIdx.x * blockDim.x + threadIdx.x;
  if (idx >= (size_t)Bc * PLANE) return;
  const size_t b = idx / PLANE, p = idx % PLANE;
  float v[NBc];
  const float* src = cmpre + (b * NBc) * PLANE + p;
  #pragma unroll
  for (int n = 0; n < NBc; ++n) v[n] = src[(size_t)n * PLANE];
  const float s0 = smooth[0], s1 = smooth[1], s2 = smooth[2],
              s3 = smooth[3], s4 = smooth[4];
  float accum = 0.0f;
  float* dst = cmOut + (b * NBc) * PLANE + p;
  #pragma unroll
  for (int n = 0; n < NBc; ++n) {
    float sm = s2 * v[n];
    if (n >= 2)       sm += s0 * v[n - 2];
    if (n >= 1)       sm += s1 * v[n - 1];
    if (n + 1 < NBc)  sm += s3 * v[n + 1];
    if (n + 2 < NBc)  sm += s4 * v[n + 2];
    dst[(size_t)n * PLANE] = sm;
    accum += sm * agg[n];
  }
  cm1[idx] = tanhf(accum);
}

// ---------------------------------------------------------------------------
// Integral-image tail. Only I(r,c) at r,c in {0..9} U {503..512} are needed.
// ---------------------------------------------------------------------------
__global__ void col_prefix_kernel(const float* __restrict__ cm1,
                                  float* __restrict__ svals)
{
  const int t = blockIdx.x * blockDim.x + threadIdx.x;  // b*512 + c
  if (t >= Bc * Lc) return;
  const int b = t >> 9, c = t & 511;
  const float* base = cm1 + (size_t)b * Lc * Lc + c;
  float s = 0.0f;
  int p = 0;
  for (int i = 0; i <= Lc; ++i) {
    while (p < 20 && ((p < 10) ? p : (493 + p)) == i) {
      svals[((size_t)b * 20 + p) * Lc + c] = s;
      ++p;
    }
    if (i < Lc) s += base[(size_t)i * Lc];
  }
}

__global__ void row_prefix_kernel(const float* __restrict__ svals,
                                  float* __restrict__ ivals)
{
  const int t = blockIdx.x * blockDim.x + threadIdx.x;  // b*20 + p
  if (t >= Bc * 20) return;
  const int b = t / 20, p = t % 20;
  const float* row = svals + ((size_t)b * 20 + p) * Lc;
  float s = 0.0f;
  int q = 0;
  for (int c = 0; c <= Lc; ++c) {
    while (q < 20 && ((q < 10) ? q : (493 + q)) == c) {
      ivals[(b * 20 + p) * 20 + q] = s;
      ++q;
    }
    if (c < Lc) s += row[c];
  }
}

__global__ void final_prob_kernel(const float* __restrict__ ivals,
                                  const float* __restrict__ Lmat,
                                  const float* __restrict__ biasPtr,
                                  float* __restrict__ out)
{
  const int b = threadIdx.x;
  if (b >= Bc) return;
  const float* I = ivals + b * 400;
  float acc = 0.0f;
  for (int p = 0; p < PWc; ++p)
    for (int q = 0; q < PWc; ++q) {
      const float w = I[(10 + p) * 20 + (10 + q)] - I[(10 + p) * 20 + q]
                    - I[p * 20 + (10 + q)]        + I[p * 20 + q];
      acc += tanhf(w) * Lmat[p * PWc + q];
    }
  acc += biasPtr[0];
  out[b] = 1.0f / (1.0f + __expf(-acc));
}

// ---------------------------------------------------------------------------
// Host orchestration
// ---------------------------------------------------------------------------
static inline ZMap zm(int div, long long s1, int mod, long long s2) {
  ZMap m; m.div = div; m.s1 = s1; m.mod = mod; m.s2 = s2; return m;
}

static void launch_gemm(hipStream_t s, const _Float16* A, const _Float16* Bm,
                        const float* bias, const float* resid,
                        float* outF, _Float16* outH,
                        int M, int N, int K, int lda, int ldb, int ldc,
                        ZMap za, ZMap zb, ZMap zc,
                        float scale, int actF, int actH, int batch)
{
  dim3 grid((N + 127) / 128, (M + 63) / 64, batch);
  wmma_gemm_kernel<<<grid, 256, 0, s>>>(A, Bm, bias, resid, outF, outH,
                                        M, N, K, lda, ldb, ldc,
                                        za, zb, zc, scale, actF, actH);
}

static void launch_cvt(hipStream_t s, const float* src, _Float16* dst, size_t n) {
  int blocks = (int)((n + 255) / 256);
  if (blocks > 8192) blocks = 8192;
  cvt_f32_f16_kernel<<<blocks, 256, 0, s>>>(src, dst, n);
}

extern "C" void kernel_launch(void* const* d_in, const int* in_sizes, int n_in,
                              void* d_out, int out_size, void* d_ws, size_t ws_size,
                              hipStream_t stream) {
  (void)in_sizes; (void)n_in; (void)out_size; (void)ws_size;
  const float* x1       = (const float*)d_in[0];
  const float* x2       = (const float*)d_in[1];
  const float* proj_w   = (const float*)d_in[2];
  const float* proj_b   = (const float*)d_in[3];
  const float* mha_in_w = (const float*)d_in[4];
  const float* mha_in_b = (const float*)d_in[5];
  const float* mha_out_w= (const float*)d_in[6];
  const float* mha_out_b= (const float*)d_in[7];
  const float* ca_in_w  = (const float*)d_in[8];
  const float* ca_in_b  = (const float*)d_in[9];
  const float* ca_out_w = (const float*)d_in[10];
  const float* ca_out_b = (const float*)d_in[11];
  const float* Wbins    = (const float*)d_in[12];
  const float* agg      = (const float*)d_in[13];
  const float* Lmat     = (const float*)d_in[14];
  const float* bias0    = (const float*)d_in[15];
  const float* smooth   = (const float*)d_in[16];

  const size_t PLANE = (size_t)Lc * Lc;
  const size_t CME   = (size_t)Bc * NBc * PLANE;   // cm element count
  float* outCm = (float*)d_out;
  float* outPp = (float*)d_out + CME;

  // ---- bump allocator over d_ws ----
  char* base = (char*)d_ws;
  size_t off = 0;
  auto alloc = [&](size_t bytes) -> char* {
    char* p = base + off;
    off = (off + bytes + 255) & ~(size_t)255;
    return p;
  };

  // persistent region
  _Float16* whproj   = (_Float16*)alloc((size_t)Dc * INc * 2);
  _Float16* whmha_in = (_Float16*)alloc((size_t)TDc * Dc * 2);
  _Float16* whmha_out= (_Float16*)alloc((size_t)Dc * Dc * 2);
  _Float16* whca_in  = (_Float16*)alloc((size_t)TDc * Dc * 2);
  _Float16* whca_out = (_Float16*)alloc((size_t)Dc * Dc * 2);
  _Float16* whbinsT  = (_Float16*)alloc((size_t)NBc * HALFc * Dc * 2);
  float*    x1f      = (float*)alloc((size_t)MBLc * Dc * 4);
  float*    x2f      = (float*)alloc((size_t)MBLc * Dc * 4);
  _Float16* x1t      = (_Float16*)alloc((size_t)MBLc * Dc * 2);
  _Float16* x2t      = (_Float16*)alloc((size_t)MBLc * Dc * 2);
  float*    cm1      = (float*)alloc((size_t)Bc * PLANE * 4);
  float*    svals    = (float*)alloc((size_t)Bc * 20 * Lc * 4);
  float*    ivals    = (float*)alloc((size_t)Bc * 400 * 4);

  // dynamic region: attention-phase buffers, later aliased by bins-phase
  const size_t dynBase = off;
  _Float16* x1h   = (_Float16*)alloc((size_t)MBLc * INc * 2);
  _Float16* x2h   = (_Float16*)alloc((size_t)MBLc * INc * 2);
  _Float16* xcur1 = (_Float16*)alloc((size_t)MBLc * Dc * 2);
  _Float16* xcur2 = (_Float16*)alloc((size_t)MBLc * Dc * 2);
  _Float16* qkv1  = (_Float16*)alloc((size_t)MBLc * TDc * 2);
  _Float16* qkv2  = (_Float16*)alloc((size_t)MBLc * TDc * 2);
  float*    Sbuf  = (float*)alloc((size_t)Bc * Hc * PLANE * 4);
  _Float16* Pbuf  = (_Float16*)alloc((size_t)Bc * Hc * PLANE * 2);
  _Float16* Obuf  = (_Float16*)alloc((size_t)MBLc * Dc * 2);
  _Float16* vtbuf = (_Float16*)alloc((size_t)MBLc * Dc * 2);   // V^T per (b,h)
  // bins-phase aliases (valid once all attention reads have drained in-order)
  off = dynBase;
  _Float16* h1    = (_Float16*)alloc((size_t)Bc * NBc * Lc * HALFc * 2);
  _Float16* h2    = (_Float16*)alloc((size_t)Bc * NBc * Lc * HALFc * 2);
  float*    cmpre = (float*)alloc(CME * 4);

  const ZMap Z0 = zm(1, 0, 1, 0);
  const ZMap Zq = zm(Hc, (long long)Lc * TDc, Hc, DHc);     // per (b,h) in qkv
  const ZMap Zs = zm(1, (long long)PLANE, 1, 0);            // per z plane
  const ZMap Zo = zm(Hc, (long long)Lc * Dc, Hc, DHc);      // per (b,h) in O
  const ZMap Zv = zm(1, (long long)DHc * Lc, 1, 0);         // per (b,h) in V^T

  // ---- weight + input casts to f16 (Wbins also transposed to [N,K]) ----
  launch_cvt(stream, proj_w,    whproj,    (size_t)Dc * INc);
  launch_cvt(stream, mha_in_w,  whmha_in,  (size_t)TDc * Dc);
  launch_cvt(stream, mha_out_w, whmha_out, (size_t)Dc * Dc);
  launch_cvt(stream, ca_in_w,   whca_in,   (size_t)TDc * Dc);
  launch_cvt(stream, ca_out_w,  whca_out,  (size_t)Dc * Dc);
  wbins_transpose_kernel<<<(unsigned)(((size_t)NBc * Dc * HALFc + 255) / 256),
                           256, 0, stream>>>(Wbins, whbinsT);
  launch_cvt(stream, x1, x1h, (size_t)MBLc * INc);
  launch_cvt(stream, x2, x2h, (size_t)MBLc * INc);

  // ---- projection: x = x @ proj_w^T + proj_b ----
  launch_gemm(stream, x1h, whproj, proj_b, nullptr, x1f, xcur1,
              MBLc, Dc, INc, INc, INc, Dc, Z0, Z0, Z0, 1.f, 0, 0, 1);
  launch_gemm(stream, x2h, whproj, proj_b, nullptr, x2f, xcur2,
              MBLc, Dc, INc, INc, INc, Dc, Z0, Z0, Z0, 1.f, 0, 0, 1);

  // ---- self-attention (x1 then x2), residual + tanh-f16 copy for CA ----
  for (int t = 0; t < 2; ++t) {
    _Float16* xc  = t ? xcur2 : xcur1;
    _Float16* qkv = t ? qkv2  : qkv1;
    float*    xf  = t ? x2f   : x1f;
    _Float16* xt  = t ? x2t   : x1t;
    launch_gemm(stream, xc, whmha_in, mha_in_b, nullptr, nullptr, qkv,
                MBLc, TDc, Dc, Dc, Dc, TDc, Z0, Z0, Z0, 1.f, 0, 0, 1);
    launch_gemm(stream, qkv, qkv + Dc, nullptr, nullptr, Sbuf, nullptr,
                Lc, Lc, DHc, TDc, TDc, Lc, Zq, Zq, Zs, 0.125f, 0, 0, Bc * Hc);
    softmax_rows_kernel<<<(Bc * Hc * Lc + 7) / 8, 256, 0, stream>>>(
        Sbuf, Pbuf, Bc * Hc * Lc, Lc);
    vtrans_kernel<<<Bc * Hc * (Lc / 32), 256, 0, stream>>>(qkv + 2 * Dc, vtbuf);
    launch_gemm(stream, Pbuf, vtbuf, nullptr, nullptr, nullptr, Obuf,
                Lc, DHc, Lc, Lc, Lc, Dc, Zs, Zv, Zo, 1.f, 0, 0, Bc * Hc);
    // out-proj: xf = mha + xf ; xt = tanh(mha + xf) (f16 input to cross-attn)
    launch_gemm(stream, Obuf, whmha_out, mha_out_b, xf, xf, xt,
                MBLc, Dc, Dc, Dc, Dc, Dc, Z0, Z0, Z0, 1.f, 0, 1, 1);
  }

  // ---- cross-attention QKV (q | kv packed at cols [0:512 | 512:1536]) ----
  launch_gemm(stream, x1t, whca_in, ca_in_b, nullptr, nullptr, qkv1,
              MBLc, Dc, Dc, Dc, Dc, TDc, Z0, Z0, Z0, 1.f, 0, 0, 1);
  launch_gemm(stream, x1t, whca_in + (size_t)Dc * Dc, ca_in_b + Dc,
              nullptr, nullptr, qkv1 + Dc,
              MBLc, 2 * Dc, Dc, Dc, Dc, TDc, Z0, Z0, Z0, 1.f, 0, 0, 1);
  launch_gemm(stream, x2t, whca_in, ca_in_b, nullptr, nullptr, qkv2,
              MBLc, Dc, Dc, Dc, Dc, TDc, Z0, Z0, Z0, 1.f, 0, 0, 1);
  launch_gemm(stream, x2t, whca_in + (size_t)Dc * Dc, ca_in_b + Dc,
              nullptr, nullptr, qkv2 + Dc,
              MBLc, 2 * Dc, Dc, Dc, Dc, TDc, Z0, Z0, Z0, 1.f, 0, 0, 1);

  // ---- CA1: q from x1, kv from x2 -> x1t = tanh(x1 + CA); CA2 symmetric ----
  for (int t = 0; t < 2; ++t) {
    _Float16* qkvQ = t ? qkv2 : qkv1;
    _Float16* qkvK = t ? qkv1 : qkv2;
    float*    xf   = t ? x2f  : x1f;
    _Float16* xt   = t ? x2t  : x1t;
    launch_gemm(stream, qkvQ, qkvK + Dc, nullptr, nullptr, Sbuf, nullptr,
                Lc, Lc, DHc, TDc, TDc, Lc, Zq, Zq, Zs, 0.125f, 0, 0, Bc * Hc);
    softmax_rows_kernel<<<(Bc * Hc * Lc + 7) / 8, 256, 0, stream>>>(
        Sbuf, Pbuf, Bc * Hc * Lc, Lc);
    vtrans_kernel<<<Bc * Hc * (Lc / 32), 256, 0, stream>>>(qkvK + 2 * Dc, vtbuf);
    launch_gemm(stream, Pbuf, vtbuf, nullptr, nullptr, nullptr, Obuf,
                Lc, DHc, Lc, Lc, Lc, Dc, Zs, Zv, Zo, 1.f, 0, 0, Bc * Hc);
    launch_gemm(stream, Obuf, whca_out, ca_out_b, xf, nullptr, xt,
                MBLc, Dc, Dc, Dc, Dc, Dc, Z0, Z0, Z0, 1.f, 0, 1, 1);
  }

  // ---- h = tanh(x @ WbinsT[n]^T) for each bin (batch over b*NBINS) ----
  const ZMap ZbinA = zm(NBc, (long long)Lc * Dc, 1, 0);
  const ZMap ZbinB = zm(1, 0, NBc, (long long)HALFc * Dc);
  const ZMap ZbinC = zm(1, (long long)Lc * HALFc, 1, 0);
  launch_gemm(stream, x1t, whbinsT, nullptr, nullptr, nullptr, h1,
              Lc, HALFc, Dc, Dc, Dc, HALFc, ZbinA, ZbinB, ZbinC,
              1.f, 0, 1, Bc * NBc);
  launch_gemm(stream, x2t, whbinsT, nullptr, nullptr, nullptr, h2,
              Lc, HALFc, Dc, Dc, Dc, HALFc, ZbinA, ZbinB, ZbinC,
              1.f, 0, 1, Bc * NBc);

  // ---- cm_pre[b,n] = h1[b,n] @ h2[b,n]^T ----
  const ZMap Zh = zm(1, (long long)Lc * HALFc, 1, 0);
  launch_gemm(stream, h1, h2, nullptr, nullptr, cmpre, nullptr,
              Lc, Lc, HALFc, HALFc, HALFc, Lc, Zh, Zh, Zs,
              1.f, 0, 0, Bc * NBc);

  // ---- fused bin smoothing -> d_out(cm), + agg reduce + tanh -> cm1 ----
  {
    const size_t total = (size_t)Bc * PLANE;
    smooth_agg_kernel<<<(unsigned)((total + 255) / 256), 256, 0, stream>>>(
        cmpre, smooth, agg, outCm, cm1);
  }

  // ---- integral-image tail -> pp_prob ----
  col_prefix_kernel<<<(Bc * Lc + 255) / 256, 256, 0, stream>>>(cm1, svals);
  row_prefix_kernel<<<1, 256, 0, stream>>>(svals, ivals);
  final_prob_kernel<<<1, 32, 0, stream>>>(ivals, Lmat, bias0, outPp);
}
